// GraphBEVNet_5832565588167
// MI455X (gfx1250) — matrix operand order repaired and checked by reading, not compiled, not run on hardware
//
#include <hip/hip_runtime.h>
#include <hip/hip_bf16.h>
#include <math.h>

#define A_N 8
#define C_N 256
#define H_N 80
#define W_N 108
#define HW_N (H_N * W_N)      // 8640
#define NC 7
#define OH 256
#define OW 256

typedef __attribute__((ext_vector_type(16))) _Float16 v16h;
typedef __attribute__((ext_vector_type(8)))  float    v8f;

// ---------------------------------------------------------------------------
// Kernel 0: column sums of w_cls  -> sw[o] = sum_k w[o,k]
// ---------------------------------------------------------------------------
__global__ void k_colsum(const float* __restrict__ w, float* __restrict__ sw) {
    int o = threadIdx.x;
    if (o < NC) {
        float s = 0.f;
        for (int k = 0; k < C_N; ++k) s += w[o * C_N + k];
        sw[o] = s;
    }
}

// ---------------------------------------------------------------------------
// Kernel 0b: pack w_cls into WMMA B-fragment layout, f16, zero-padded.
// wpack[((t*32 + lane)*16) + e] = B-frag element e of lane `lane` for k-tile t.
// B (32x16, KxN): lane n = lane&15, hi = lane>>4; element e -> K = t*32 + (e&7) + 16*(e>>3) + 8*hi.
// 8 k-tiles * 32 lanes * 16 halves = 4096 f16 (8 KB), loaded later as one v16h
// (2x global_load_b128) per wave per k-tile -- no predication, no cvt.
// ---------------------------------------------------------------------------
__global__ void k_packw(const float* __restrict__ w, _Float16* __restrict__ wpack) {
    int idx = blockIdx.x * blockDim.x + threadIdx.x;   // 0..4095
    if (idx >= 8 * 32 * 16) return;
    int e    = idx & 15;
    int lane = (idx >> 4) & 31;
    int t    = idx >> 9;
    int n  = lane & 15;
    int hi = lane >> 4;
    int k  = t * 32 + (e & 7) + ((e >> 3) << 4) + 8 * hi;
    wpack[idx] = (n < NC) ? (_Float16)w[n * C_N + k] : (_Float16)0.0f;
}

// ---------------------------------------------------------------------------
// Kernel 2 (WMMA): z[a,o,p] = sum_k w[o,k]*x[a,k,p] + sw[o]*cm[a,p]
// Per wave: one 16-pixel x 16-class tile, K=256 via 8x v_wmma_f32_16x16x32_f16.
// 8640 px = 540 tiles/agent = 135 blocks * 4 waves (no partial waves; EXEC full
// for every WMMA -- all control flow is wave-uniform before the WMMAs).
// ---------------------------------------------------------------------------
__global__ __launch_bounds__(128)
void k_gemm_wmma(const float* __restrict__ x, const _Float16* __restrict__ wpack,
                 const float* __restrict__ cm, const float* __restrict__ sw,
                 float* __restrict__ z) {
    int wave = threadIdx.x >> 5;
    int lane = threadIdx.x & 31;
    int tile = blockIdx.x * 4 + wave;        // 0..539
    int a    = blockIdx.y;
    int m    = lane & 15;                    // A row (pixel) / D col (class)
    int hi   = lane >> 4;
    int p0   = tile * 16;
    const float* xa = x + (size_t)a * C_N * HW_N;
    const v16h*  wp = (const v16h*)wpack;    // [8 ktiles][32 lanes]

    v8f acc = {};
#pragma unroll
    for (int t = 0; t < 8; ++t) {
        int k0 = t * 32;
        // A fragment (16x32, M=pixels, K=channels): lane m, element e -> K
        v16h aF;
#pragma unroll
        for (int e = 0; e < 16; ++e) {
            int k = k0 + (e & 7) + ((e >> 3) << 4) + 8 * hi;
            aF[e] = (_Float16)xa[k * HW_N + p0 + m];
        }
        // B fragment: single aligned 32B vector load from the packed table
        v16h bF = wp[t * 32 + lane];
        acc = __builtin_amdgcn_wmma_f32_16x16x32_f16(
            /*neg_a=*/false, aF, /*neg_b=*/false, bF,
            /*c_mod=*/(short)0, acc, /*reuse_a=*/false, /*reuse_b=*/false);
    }

    // D layout: VGPR r -> row (pixel) r + 8*hi, col (class) m. 8 consecutive px.
    if (m < NC) {
        float s = sw[m];
        float*       zp  = z  + ((size_t)(a * NC + m)) * HW_N + p0 + 8 * hi;
        const float* cmp = cm + (size_t)a * HW_N + p0 + 8 * hi;
#pragma unroll
        for (int r = 0; r < 8; ++r)
            zp[r] = acc[r] + s * cmp[r];
    }
}

// ---------------------------------------------------------------------------
// Kernel 1: car_masks [8,256,256] -> cm [8,80,108]
// jax.image.resize(method="bilinear") == antialiased separable triangle filter
// ---------------------------------------------------------------------------
__global__ void k_resize_cm(const float* __restrict__ cmask, float* __restrict__ cm) {
    int idx = blockIdx.x * blockDim.x + threadIdx.x;
    if (idx >= A_N * HW_N) return;
    int a = idx / HW_N;
    int p = idx - a * HW_N;
    int oy = p / W_N, ox = p - oy * W_N;

    const float ry = 256.0f / (float)H_N;   // support radius, 3.2
    const float rx = 256.0f / (float)W_N;   // ~2.370
    const float ky = (float)H_N / 256.0f;   // distance scale 0.3125
    const float kx = (float)W_N / 256.0f;   // 0.421875

    float fy = (oy + 0.5f) * ry - 0.5f;
    float fx = (ox + 0.5f) * rx - 0.5f;
    int jy0 = (int)ceilf(fy - ry);  if (jy0 < 0)   jy0 = 0;
    int jy1 = (int)floorf(fy + ry); if (jy1 > 255) jy1 = 255;
    int jx0 = (int)ceilf(fx - rx);  if (jx0 < 0)   jx0 = 0;
    int jx1 = (int)floorf(fx + rx); if (jx1 > 255) jx1 = 255;

    float acc = 0.f, wsum = 0.f;
    for (int jy = jy0; jy <= jy1; ++jy) {
        float wy = 1.0f - fabsf((float)jy - fy) * ky; if (wy < 0.f) wy = 0.f;
        float rowacc = 0.f, rowwsum = 0.f;
        for (int jx = jx0; jx <= jx1; ++jx) {
            float wx = 1.0f - fabsf((float)jx - fx) * kx; if (wx < 0.f) wx = 0.f;
            rowacc  += wx * cmask[(a * 256 + jy) * 256 + jx];
            rowwsum += wx;
        }
        acc  += wy * rowacc;
        wsum += wy * rowwsum;
    }
    cm[idx] = acc / wsum;
}

// ---------------------------------------------------------------------------
// Kernel 3: solo probabilities  ps = sigmoid(z + b)
// ---------------------------------------------------------------------------
__global__ void k_solo(const float* __restrict__ z, const float* __restrict__ bias,
                       float* __restrict__ ps) {
    int idx = blockIdx.x * blockDim.x + threadIdx.x;
    if (idx >= A_N * NC * HW_N) return;
    int o = (idx / HW_N) % NC;
    float v = z[idx] + bias[o];
    ps[idx] = 1.0f / (1.0f + __expf(-v));
}

// ---------------------------------------------------------------------------
// Kernel 4: aggregate = sigmoid( b + sum_j adj[i,j] * warp(z_j, M[i,j]) )
// (warp commuted past the 1x1 conv: only 7 channels to sample)
// ---------------------------------------------------------------------------
__global__ void k_aggr(const float* __restrict__ z, const int* __restrict__ adj,
                       const float* __restrict__ rel, const float* __restrict__ bias,
                       float* __restrict__ pa) {
    int idx = blockIdx.x * blockDim.x + threadIdx.x;
    if (idx >= A_N * HW_N) return;
    int i = idx / HW_N;
    int p = idx - i * HW_N;
    int v = p / W_N, u = p - v * W_N;

    float acc[NC];
#pragma unroll
    for (int o = 0; o < NC; ++o) acc[o] = bias[o];

    for (int j = 0; j < A_N; ++j) {
        if (adj[i * A_N + j] == 0) continue;
        const float* M = rel + (size_t)(i * A_N + j) * 6;
        float sx = M[0] * (float)u + M[1] * (float)v + M[2];
        float sy = M[3] * (float)u + M[4] * (float)v + M[5];
        float fx0 = floorf(sx), fy0 = floorf(sy);
        float wx = sx - fx0, wy = sy - fy0;
        int x0 = (int)fx0, y0 = (int)fy0;
        const float* zj = z + (size_t)j * NC * HW_N;

        float tw[4] = {(1.f - wy) * (1.f - wx), (1.f - wy) * wx,
                       wy * (1.f - wx),          wy * wx};
        int xs[4] = {x0, x0 + 1, x0,     x0 + 1};
        int ys[4] = {y0, y0,     y0 + 1, y0 + 1};
#pragma unroll
        for (int t = 0; t < 4; ++t) {
            int xx = xs[t], yy = ys[t];
            bool valid = (xx >= 0) && (xx < W_N) && (yy >= 0) && (yy < H_N);
            float wt = valid ? tw[t] : 0.f;
            if (wt != 0.f) {
                int xc = xx < 0 ? 0 : (xx > W_N - 1 ? W_N - 1 : xx);
                int yc = yy < 0 ? 0 : (yy > H_N - 1 ? H_N - 1 : yy);
                int off = yc * W_N + xc;
#pragma unroll
                for (int o = 0; o < NC; ++o)
                    acc[o] += wt * zj[o * HW_N + off];
            }
        }
    }
#pragma unroll
    for (int o = 0; o < NC; ++o)
        pa[(i * NC + o) * HW_N + p] = 1.0f / (1.0f + __expf(-acc[o]));
}

// ---------------------------------------------------------------------------
// Kernel 5: align_corners=True bilinear resize [*,80,108] -> [*,256,256]
// for both halves of the output (solo first, then aggr).
// ---------------------------------------------------------------------------
__global__ void k_resize_out(const float* __restrict__ ps, const float* __restrict__ pa,
                             float* __restrict__ out) {
    long long idx = (long long)blockIdx.x * blockDim.x + threadIdx.x;
    const long long half = (long long)A_N * NC * OH * OW;
    if (idx >= 2 * half) return;
    const float* src = (idx < half) ? ps : pa;
    long long r = (idx < half) ? idx : idx - half;
    int ox = (int)(r % OW);
    int oy = (int)((r / OW) % OH);
    int ao = (int)(r / ((long long)OW * OH));   // a*NC + o

    float ysf = (float)oy * (float)(H_N - 1) / (float)(OH - 1);
    float xsf = (float)ox * (float)(W_N - 1) / (float)(OW - 1);
    int y0 = (int)floorf(ysf); if (y0 > H_N - 2) y0 = H_N - 2; if (y0 < 0) y0 = 0;
    int x0 = (int)floorf(xsf); if (x0 > W_N - 2) x0 = W_N - 2; if (x0 < 0) x0 = 0;
    float wy = ysf - (float)y0, wx = xsf - (float)x0;

    const float* s = src + (long long)ao * HW_N;
    float v00 = s[y0 * W_N + x0],       v01 = s[y0 * W_N + x0 + 1];
    float v10 = s[(y0 + 1) * W_N + x0], v11 = s[(y0 + 1) * W_N + x0 + 1];
    out[idx] = (v00 * (1.f - wx) + v01 * wx) * (1.f - wy)
             + (v10 * (1.f - wx) + v11 * wx) * wy;
}

// ---------------------------------------------------------------------------
// Workspace layout (floats):
//   cm    : [8,8640]          @ 0         (69120)
//   sw    : [8]               @ 69120     (8)
//   z     : [8,7,8640]        @ 69128     (483840)
//   ps    : [8,7,8640]        @ 552968    (483840)
//   pa    : [8,7,8640]        @ 1036808   (483840)
//   wpack : [4096] f16        @ 1520648   (2048 floats; 32B-aligned)
// total ~6.09 MB
// ---------------------------------------------------------------------------
extern "C" void kernel_launch(void* const* d_in, const int* in_sizes, int n_in,
                              void* d_out, int out_size, void* d_ws, size_t ws_size,
                              hipStream_t stream) {
    const float* x     = (const float*)d_in[0];
    const float* rel   = (const float*)d_in[1];
    const int*   adj   = (const int*)d_in[2];
    const float* cmask = (const float*)d_in[3];
    const float* wcls  = (const float*)d_in[4];
    const float* bcls  = (const float*)d_in[5];

    float* ws = (float*)d_ws;
    float* cm = ws;
    float* sw = ws + 69120;
    float* z  = ws + 69128;
    float* ps = ws + 552968;
    float* pa = ws + 1036808;
    _Float16* wpack = (_Float16*)(ws + 1520648);
    float* out = (float*)d_out;

    k_colsum<<<1, 32, 0, stream>>>(wcls, sw);
    k_packw<<<16, 256, 0, stream>>>(wcls, wpack);

    { int n = A_N * HW_N;
      k_resize_cm<<<(n + 255) / 256, 256, 0, stream>>>(cmask, cm); }

    { dim3 grid(135, A_N);           // 135 blocks * 4 waves = 540 tiles/agent
      k_gemm_wmma<<<grid, 128, 0, stream>>>(x, wpack, cm, sw, z); }

    { int n = A_N * NC * HW_N;
      k_solo<<<(n + 255) / 256, 256, 0, stream>>>(z, bcls, ps); }

    { int n = A_N * HW_N;
      k_aggr<<<(n + 255) / 256, 256, 0, stream>>>(z, adj, rel, bcls, pa); }

    { long long n = 2LL * A_N * NC * OH * OW;
      k_resize_out<<<(int)((n + 255) / 256), 256, 0, stream>>>(ps, pa, out); }
}
